// RVQVectorQuantizer2_75909251989932
// MI455X (gfx1250) — matrix operand order repaired and checked by reading, not compiled, not run on hardware
//
#include <hip/hip_runtime.h>

// ---------------------------------------------------------------------------
// RVQ (residual vector quantizer), MI455X / gfx1250, wave32.
// N=32768 rows, D=64, K=1024 codes, L=4 layers.
// GEMM = 17.2 GFLOP -> matrix cores. Precision: bf16 hi/lo split, 3x
// v_wmma_f32_16x16x32_bf16 per K-step => ~fp32-accurate argmin at bf16 rates.
// argmin(||e||^2 - 2 sim) == argmax(sim - 0.5||e||^2). C operand starts as
// the free inline-0 literal; the -0.5||e||^2 bias is added post-WMMA from a
// transposed bias table ([lane][tile], one float4 per 4 tiles out of padded
// LDS) so nothing stalls the WMMA issue chain. Inner sweep unrolled 4x.
// ---------------------------------------------------------------------------

typedef __bf16 bf16t;
typedef __attribute__((ext_vector_type(16))) bf16t v16bf;
typedef __attribute__((ext_vector_type(8)))  float v8f;

#define DDIM    64
#define KCODES  1024
#define LLAYERS 4
#define NROWS   32768

union U16BF { unsigned int u[8]; v16bf v; };

__device__ __forceinline__ unsigned int bf16rn(float x) {
  unsigned int u = __float_as_uint(x);
  return (u + 0x7FFFu + ((u >> 16) & 1u)) >> 16;   // round-to-nearest-even
}
__device__ __forceinline__ float bf16tof(unsigned int b) {
  return __uint_as_float(b << 16);
}
__device__ __forceinline__ void split2(float f0, float f1,
                                       unsigned int& hi, unsigned int& lo) {
  unsigned int h0 = bf16rn(f0), h1 = bf16rn(f1);
  unsigned int l0 = bf16rn(f0 - bf16tof(h0));
  unsigned int l1 = bf16rn(f1 - bf16tof(h1));
  hi = h0 | (h1 << 16);
  lo = l0 | (l1 << 16);
}

// Load 16 contiguous bf16 (32B) into a v16bf fragment (two b128 loads).
__device__ __forceinline__ v16bf loadB16(const unsigned short* p) {
  U16BF r;
  const int4* q = reinterpret_cast<const int4*>(p);
  int4 a = q[0], b = q[1];
  r.u[0] = (unsigned)a.x; r.u[1] = (unsigned)a.y;
  r.u[2] = (unsigned)a.z; r.u[3] = (unsigned)a.w;
  r.u[4] = (unsigned)b.x; r.u[5] = (unsigned)b.y;
  r.u[6] = (unsigned)b.z; r.u[7] = (unsigned)b.w;
  return r.v;
}

// ---------------------------------------------------------------------------
// Prep: repack embeddings [D,K,L] (d-stride 4096 floats!) into WMMA-friendly
// [L][K][D] bf16 hi/lo + fp32 copy + transposed bias table
// ebiasT[l][k%16][k/16] = -0.5*||e||^2  (per-lane-contiguous in the sweep).
// ---------------------------------------------------------------------------
__global__ void rvq_prep(const float* __restrict__ emb,
                         unsigned short* __restrict__ bhi,
                         unsigned short* __restrict__ blo,
                         float* __restrict__ embT,
                         float* __restrict__ ebiasT) {
  const int tid = blockIdx.x * blockDim.x + threadIdx.x;   // 0..4095
  if (tid >= KCODES * LLAYERS) return;
  const int l = tid & (LLAYERS - 1);
  const int k = tid >> 2;
  const unsigned ob = (unsigned)(l * KCODES + k) * DDIM;
  float nrm = 0.0f;
  for (int d = 0; d < DDIM; ++d) {
    float v = emb[(unsigned)d * (KCODES * LLAYERS) + (unsigned)tid];
    nrm += v * v;
    unsigned int h = bf16rn(v);
    unsigned int lo = bf16rn(v - bf16tof(h));
    bhi[ob + d]  = (unsigned short)h;
    blo[ob + d]  = (unsigned short)lo;
    embT[ob + d] = v;
  }
  // transposed: lane m = k%16 owns codes {t*16+m}, contiguous over t = k/16
  ebiasT[l * KCODES + (k & 15) * 64 + (k >> 4)] = -0.5f * nrm;
}

// ---------------------------------------------------------------------------
// Main: 8 waves/block, 16 rows/wave, 128 rows/block, 256 blocks.
// ---------------------------------------------------------------------------
__global__ __launch_bounds__(256)
void rvq_main(const float* __restrict__ x,
              const unsigned short* __restrict__ bhi,
              const unsigned short* __restrict__ blo,
              const float* __restrict__ embT,
              const float* __restrict__ ebiasT,
              float* __restrict__ out) {
  __shared__ float res[8][16][68];     // padded: kills bank conflicts
  __shared__ float ebS[16][68];        // [lane m][tile t], padded
  __shared__ int   idxS[8][16];

  const int tid  = threadIdx.x;
  const int wv   = tid >> 5;
  const int lane = tid & 31;
  const int hf   = lane >> 4;          // half of wave (K/N-interleave select)
  const int m    = lane & 15;          // row (A/M) == column (B/N) index
  const int waveRow0 = blockIdx.x * 128 + wv * 16;

  // ---- load x tile -> residual (LDS) ----
  {
    const float4* xs =
        reinterpret_cast<const float4*>(x + (size_t)(waveRow0 + m) * DDIM + hf * 32);
    float4* rd = reinterpret_cast<float4*>(&res[wv][m][hf * 32]);
#pragma unroll
    for (int i = 0; i < 8; ++i) rd[i] = xs[i];
  }

  for (int ll = 0; ll < LLAYERS; ++ll) {
    // per-layer transposed biases -> padded LDS
    for (int i = tid; i < KCODES; i += 256)
      ebS[i >> 6][i & 63] = ebiasT[ll * KCODES + i];
    __syncthreads();

    // ---- A fragments: residual -> bf16 hi/lo in 16x32 A layout (contiguous k, same order as B) ----
    // A element e (0..15) of step s: d = s*32 + hf*16 + e
    v16bf ah0, ah1, al0, al1;
#pragma unroll
    for (int s = 0; s < 2; ++s) {
      U16BF H, L;
#pragma unroll
      for (int c = 0; c < 2; ++c) {
        const int dbase = s * 32 + hf * 16 + c * 8;
        const float4* q = reinterpret_cast<const float4*>(&res[wv][m][dbase]);
        float4 q0 = q[0], q1 = q[1];
        split2(q0.x, q0.y, H.u[c * 4 + 0], L.u[c * 4 + 0]);
        split2(q0.z, q0.w, H.u[c * 4 + 1], L.u[c * 4 + 1]);
        split2(q1.x, q1.y, H.u[c * 4 + 2], L.u[c * 4 + 2]);
        split2(q1.z, q1.w, H.u[c * 4 + 3], L.u[c * 4 + 3]);
      }
      if (s == 0) { ah0 = H.v; al0 = L.v; } else { ah1 = H.v; al1 = L.v; }
    }

    float bs[8];                       // best (sim - 0.5||e||^2), maximize
    int   bi[8];
#pragma unroll
    for (int r = 0; r < 8; ++r) { bs[r] = -3.4e38f; bi[r] = 0; }

    const unsigned short* bhiL = bhi + (size_t)ll * KCODES * DDIM;
    const unsigned short* bloL = blo + (size_t)ll * KCODES * DDIM;

    // ---- sweep 64 code tiles, 4 per bias-float4 chunk ----
    for (int tb = 0; tb < 16; ++tb) {
      const float4 b4 =
          *reinterpret_cast<const float4*>(&ebS[m][tb * 4]);   // 4 tile biases
      const float biasj[4] = {b4.x, b4.y, b4.z, b4.w};
#pragma unroll
      for (int j = 0; j < 4; ++j) {
        const int t = tb * 4 + j;
        const int code = t * 16 + m;
        const size_t ro = (size_t)code * DDIM;
        // B element e of step s: d = s*32 + hf*16 + e  -> 32B contiguous
        v16bf bh_0 = loadB16(bhiL + ro + hf * 16);
        v16bf bh_1 = loadB16(bhiL + ro + 32 + hf * 16);
        v16bf bl_0 = loadB16(bloL + ro + hf * 16);
        v16bf bl_1 = loadB16(bloL + ro + 32 + hf * 16);

        v8f acc = {0.f, 0.f, 0.f, 0.f, 0.f, 0.f, 0.f, 0.f};   // inline-0 C
        acc = __builtin_amdgcn_wmma_f32_16x16x32_bf16(false, ah0, false, bh_0,
                                                      (short)0, acc, false, false);
        acc = __builtin_amdgcn_wmma_f32_16x16x32_bf16(false, ah1, false, bh_1,
                                                      (short)0, acc, false, false);
        acc = __builtin_amdgcn_wmma_f32_16x16x32_bf16(false, ah0, false, bl_0,
                                                      (short)0, acc, false, false);
        acc = __builtin_amdgcn_wmma_f32_16x16x32_bf16(false, ah1, false, bl_1,
                                                      (short)0, acc, false, false);
        acc = __builtin_amdgcn_wmma_f32_16x16x32_bf16(false, al0, false, bh_0,
                                                      (short)0, acc, false, false);
        acc = __builtin_amdgcn_wmma_f32_16x16x32_bf16(false, al1, false, bh_1,
                                                      (short)0, acc, false, false);

#pragma unroll
        for (int r = 0; r < 8; ++r) {
          float sc = acc[r] + biasj[j];
          if (sc > bs[r]) { bs[r] = sc; bi[r] = code; }
        }
      }
    }

    // ---- cross-lane argmax within each 16-lane half (wave32 shuffles) ----
#pragma unroll
    for (int off = 8; off > 0; off >>= 1) {
#pragma unroll
      for (int r = 0; r < 8; ++r) {
        float os = __shfl_xor(bs[r], off, 32);
        int   oi = __shfl_xor(bi[r], off, 32);
        if (os > bs[r] || (os == bs[r] && oi < bi[r])) { bs[r] = os; bi[r] = oi; }
      }
    }
    if (m == 0) {        // lanes 0 and 16: rows hf*8 + r  (C layout M = r + 8*hf)
#pragma unroll
      for (int r = 0; r < 8; ++r) idxS[wv][hf * 8 + r] = bi[r];
    }
    __syncthreads();

    // ---- gather winning code (contiguous fp32 row) & update residual ----
    {
      const int code = idxS[wv][m];
      const float4* e4 = reinterpret_cast<const float4*>(
          embT + ((size_t)ll * KCODES + code) * DDIM + hf * 32);
      float4* r4 = reinterpret_cast<float4*>(&res[wv][m][hf * 32]);
#pragma unroll
      for (int i = 0; i < 8; ++i) {
        float4 ev = e4[i];
        float4 rv = r4[i];
        rv.x -= ev.x; rv.y -= ev.y; rv.z -= ev.z; rv.w -= ev.w;
        r4[i] = rv;
      }
    }
    __syncthreads();
  }

  // ---- out = x - residual_final  (== accumulated quantization) ----
  {
    const size_t rb = (size_t)(waveRow0 + m) * DDIM + hf * 32;
    const float4* xs = reinterpret_cast<const float4*>(x + rb);
    const float4* r4 = reinterpret_cast<const float4*>(&res[wv][m][hf * 32]);
    float4* od = reinterpret_cast<float4*>(out + rb);
#pragma unroll
    for (int i = 0; i < 8; ++i) {
      float4 xv = xs[i], rv = r4[i], ov;
      ov.x = xv.x - rv.x; ov.y = xv.y - rv.y;
      ov.z = xv.z - rv.z; ov.w = xv.w - rv.w;
      od[i] = ov;
    }
  }
}

// ---------------------------------------------------------------------------
extern "C" void kernel_launch(void* const* d_in, const int* in_sizes, int n_in,
                              void* d_out, int out_size, void* d_ws, size_t ws_size,
                              hipStream_t stream) {
  (void)in_sizes; (void)n_in; (void)out_size; (void)ws_size;
  const float* x   = (const float*)d_in[0];        // [32,32,32,64] fp32
  const float* emb = (const float*)d_in[1];        // [64,1024,4]  fp32
  float* out       = (float*)d_out;                // [32,32,32,64] fp32

  // workspace layout (all 256B-aligned): hi bf16 | lo bf16 | fp32 copy | bias
  char* ws = (char*)d_ws;
  unsigned short* bhi = (unsigned short*)(ws);
  unsigned short* blo = (unsigned short*)(ws + 512 * 1024);
  float* embT   = (float*)(ws + 1024 * 1024);
  float* ebiasT = (float*)(ws + 2048 * 1024);

  rvq_prep<<<16, 256, 0, stream>>>(emb, bhi, blo, embT, ebiasT);
  rvq_main<<<NROWS / 128, 256, 0, stream>>>(x, bhi, blo, embT, ebiasT, out);
}
